// SelfAttention123_52553219833949
// MI455X (gfx1250) — matrix operand order repaired and checked by reading, not compiled, not run on hardware
//
#include <hip/hip_runtime.h>
#include <hip/hip_bf16.h>

// Problem constants (from reference)
#define Bsz 2
#define Sq  2048
#define Hd  1024
#define NHd 16
#define HDd 64
#define Md  (Bsz*Sq)      // 4096 rows of x
#define LSCALE 4.0f       // lora_alpha / r

#define KSTEP 32
#define XROWS 128
#define WROWS 64
#define LROW  40          // padded LDS row stride in halves (80B = 20 banks)

typedef _Float16 f16;
typedef __attribute__((ext_vector_type(16))) _Float16 v16h;
typedef __attribute__((ext_vector_type(8)))  _Float16 v8h;
typedef __attribute__((ext_vector_type(8)))  float    v8f;

// CDNA5 async global->LDS copy (ASYNCcnt-tracked). VDST = LDS byte address,
// VADDR = 64-bit global address, GV mode (SADDR = off).
static __device__ __forceinline__ void async_b128(unsigned int lds_off, const f16* g) {
  asm volatile("global_load_async_to_lds_b128 %0, %1, off"
               :: "v"(lds_off), "v"(g)
               : "memory");
}

// Build a 16-bit A-fragment (16x32, MxK) in contiguous K order (B uses the same):
// lane L holds row M=L%16; element i -> K = 16*(L/16) + i
static __device__ __forceinline__ v16h make_frag(v8h lo, v8h hi) {
  v16h a;
#pragma unroll
  for (int i = 0; i < 8; ++i) { a[i] = lo[i]; a[i + 8] = hi[i]; }
  return a;
}

static __device__ __forceinline__ v16h load_afrag_g(const f16* rowbase, int lane) {
  const int khalf = (lane >> 4) * 16;
  v8h lo = *(const v8h*)(rowbase + khalf);
  v8h hi = *(const v8h*)(rowbase + khalf + 8);
  return make_frag(lo, hi);
}

// ---------------- conversion kernels ----------------
__global__ __launch_bounds__(256) void cvt_x_kernel(const float* __restrict__ x,
                                                    f16* __restrict__ xh, int n) {
  int i = blockIdx.x * 256 + threadIdx.x;
  if (i < n) xh[i] = (f16)x[i];
}

__global__ __launch_bounds__(256) void cvt_w_kernel(const float* __restrict__ wq,
                                                    const float* __restrict__ wk,
                                                    const float* __restrict__ wv,
                                                    f16* __restrict__ wh) {
  size_t i = (size_t)blockIdx.x * 256 + threadIdx.x;   // [0, 3*H*H)
  const size_t per = (size_t)Hd * Hd;
  const float* src = (i < per) ? wq : (i < 2 * per) ? wk : wv;
  wh[i] = (f16)src[i % per];
}

// LoRA intermediate: T[p][m][r] = sum_k x[m,k] * A_p[r,k]   (fp32, tiny)
__global__ __launch_bounds__(256) void lora_t_kernel(const float* __restrict__ x,
                                                     const float* __restrict__ Aq,
                                                     const float* __restrict__ Ak,
                                                     const float* __restrict__ Av,
                                                     float* __restrict__ T) {
  int tid = blockIdx.x * 256 + threadIdx.x;      // 3*4096*4 = 49152 total
  int r = tid & 3;
  int m = (tid >> 2) & (Md - 1);
  int p = tid >> 14;
  const float* A  = (p == 0) ? Aq : (p == 1) ? Ak : Av;
  const float* xr = x + (size_t)m * Hd;
  const float* ar = A + (size_t)r * Hd;
  float acc = 0.f;
#pragma unroll 4
  for (int k = 0; k < Hd; ++k) acc += xr[k] * ar[k];
  T[tid] = acc;                                  // layout p*Md*4 + m*4 + r
}

// ---------------- fused QKV + LoRA GEMM ----------------
// grid (N/64=16, M/128=32, 3 projections), block 256 = 8 waves.
// X[128x32] and W[64x32] tiles staged in LDS via async b128 copies, double-buffered.
// Wave w computes rows [m0+16w, +16) x cols [n0, n0+64): 4 WMMA accumulators.
__global__ __launch_bounds__(256)
void qkv_gemm_kernel(const f16* __restrict__ Xh, const f16* __restrict__ Wh,
                     const float* __restrict__ T,
                     const float* __restrict__ bq, const float* __restrict__ bk,
                     const float* __restrict__ bv,
                     const float* __restrict__ Bq, const float* __restrict__ Bk,
                     const float* __restrict__ Bv,
                     f16* __restrict__ qh, f16* __restrict__ kh, f16* __restrict__ vT) {
  __shared__ __align__(16) f16 xbuf[2 * XROWS * LROW];   // 20480 B
  __shared__ __align__(16) f16 wbuf[2 * WROWS * LROW];   // 10240 B

  const int p    = blockIdx.z;
  const int n0   = blockIdx.x * 64;
  const int m0   = blockIdx.y * 128;
  const int t    = threadIdx.x;
  const int lane = t & 31;
  const int w    = t >> 5;

  const int colL  = lane & 15;
  const int bk16  = (lane >> 4) * 16;
  const int khalf = (lane >> 4) * 8;
  const int rhalf = 8 * (lane >> 4);

  // Async-copy work assignment: X tile = 512 b128 chunks (2/thread), W tile = 256 (1/thread)
  const int xr0 = t >> 2;                 // rows 0..63
  const int xr1 = (t + 256) >> 2;         // rows 64..127
  const int xo  = (t & 3) * 8;            // half-offset within row: 0,8,16,24
  const f16* xg0 = Xh + (size_t)(m0 + xr0) * Hd + xo;
  const f16* xg1 = Xh + (size_t)(m0 + xr1) * Hd + xo;
  const f16* wg  = Wh + (size_t)p * Hd * Hd + (size_t)(n0 + xr0) * Hd + xo;

  const unsigned int xbase = (unsigned int)(uintptr_t)&xbuf[0];
  const unsigned int wbase = (unsigned int)(uintptr_t)&wbuf[0];

  v8f acc[4] = {};

  // prologue: fill buffer 0 for kk=0
  async_b128(xbase + (unsigned)((0 * XROWS + xr0) * LROW + xo) * 2, xg0);
  async_b128(xbase + (unsigned)((0 * XROWS + xr1) * LROW + xo) * 2, xg1);
  async_b128(wbase + (unsigned)((0 * WROWS + xr0) * LROW + xo) * 2, wg);

  int buf = 0;
  for (int kk = 0; kk < Hd; kk += KSTEP) {
    if (kk + KSTEP < Hd) {
      const int nb = buf ^ 1;
      async_b128(xbase + (unsigned)((nb * XROWS + xr0) * LROW + xo) * 2, xg0 + kk + KSTEP);
      async_b128(xbase + (unsigned)((nb * XROWS + xr1) * LROW + xo) * 2, xg1 + kk + KSTEP);
      async_b128(wbase + (unsigned)((nb * WROWS + xr0) * LROW + xo) * 2, wg + kk + KSTEP);
      asm volatile("s_wait_asynccnt 0x3" ::: "memory");   // current buffer's 3 done
    } else {
      asm volatile("s_wait_asynccnt 0x0" ::: "memory");
    }
    __syncthreads();   // all waves' deposits visible

    // A fragment from LDS: wave-local rows w*16 + colL
    const f16* arow = &xbuf[(buf * XROWS + w * 16 + colL) * LROW];
    v8h alo = *(const v8h*)(arow + bk16);
    v8h ahi = *(const v8h*)(arow + bk16 + 8);
    v16h a = make_frag(alo, ahi);

#pragma unroll
    for (int nt = 0; nt < 4; ++nt) {
      // B fragment (32x16 KxN): lane holds col N=L%16, 16 contiguous K at bk16
      const f16* brow = &wbuf[(buf * WROWS + nt * 16 + colL) * LROW + bk16];
      v8h blo = *(const v8h*)brow;
      v8h bhi = *(const v8h*)(brow + 8);
      v16h b = make_frag(blo, bhi);   // contiguous 16 halves
      acc[nt] = __builtin_amdgcn_wmma_f32_16x16x32_f16(false, a, false, b,
                                                       (short)0, acc[nt], false, false);
    }
    __syncthreads();   // done reading buf before it is refilled next iteration
    buf ^= 1;
  }

  const float* bias = (p == 0) ? bq : (p == 1) ? bk : bv;
  const float* Bm   = (p == 0) ? Bq : (p == 1) ? Bk : Bv;
  const int wm = m0 + w * 16;

#pragma unroll
  for (int nt = 0; nt < 4; ++nt) {
    const int n  = n0 + nt * 16 + colL;
    const float bn = bias[n];
    const float l0 = Bm[n * 4 + 0], l1 = Bm[n * 4 + 1];
    const float l2 = Bm[n * 4 + 2], l3 = Bm[n * 4 + 3];
    const int hh = n >> 6, d = n & 63;
#pragma unroll
    for (int r = 0; r < 8; ++r) {
      const int m = wm + r + rhalf;
      const float* tm = T + ((size_t)p * Md + m) * 4;
      const float lora = tm[0] * l0 + tm[1] * l1 + tm[2] * l2 + tm[3] * l3;
      const float val = acc[nt][r] + bn + LSCALE * lora;
      const int bb = m >> 11, s = m & (Sq - 1);
      if (p == 2) {
        // v stored transposed [b,h,d,s] so attention's ctx B-fragment is contiguous
        vT[(((size_t)bb * NHd + hh) * HDd + d) * Sq + s] = (f16)val;
      } else {
        f16* dst = (p == 0) ? qh : kh;
        dst[(((size_t)bb * NHd + hh) * Sq + s) * HDd + d] = (f16)val;
      }
    }
  }
}

// ---------------- flash-style causal attention ----------------
// grid (S/64=32, NH=16, B=2), block 128 = 4 waves; wave handles one 16-query tile,
// streams 32 keys per step with online softmax; P tile goes C-layout -> A-layout via LDS.
__global__ __launch_bounds__(128)
void attn_kernel(const f16* __restrict__ qh, const f16* __restrict__ kh,
                 const f16* __restrict__ vT, const float* __restrict__ mask,
                 float* __restrict__ out) {
  __shared__ __align__(16) f16 plds[4 * 16 * 40];   // per-wave 16x(32+8pad) tiles
  const int lane = threadIdx.x & 31;
  const int w    = threadIdx.x >> 5;
  const int b    = blockIdx.z;
  const int h    = blockIdx.y;
  const int q0   = (blockIdx.x * 4 + w) * 16;

  const f16* qp = qh + (size_t)(b * NHd + h) * Sq * HDd;
  const f16* kp = kh + (size_t)(b * NHd + h) * Sq * HDd;
  const f16* vp = vT + (size_t)(b * NHd + h) * HDd * Sq;
  const float* mrow = mask + (size_t)b * Sq;

  const int colL  = lane & 15;
  const int rhalf = 8 * (lane >> 4);
  const int bk16  = (lane >> 4) * 16;
  f16* myld = plds + w * 16 * 40;

  // Q fragments for K(d)=0..31 and 32..63
  const int qrow = q0 + colL;
  v16h qf0 = load_afrag_g(qp + (size_t)qrow * HDd + 0,  lane);
  v16h qf1 = load_afrag_g(qp + (size_t)qrow * HDd + 32, lane);

  float rmax[8], rsum[8];
#pragma unroll
  for (int r = 0; r < 8; ++r) { rmax[r] = -1e30f; rsum[r] = 0.f; }
  v8f accO[4] = {};

  for (int k0 = 0; k0 < q0 + 16; k0 += 32) {
    // scores for 32 keys: two 16x16 tiles, K(d)=64 in two WMMA steps
    v8f s0 = {}, s1 = {};
    {
      v16h kf00 = *(const v16h*)(kp + (size_t)(k0 + colL) * HDd + 0  + bk16);
      v16h kf01 = *(const v16h*)(kp + (size_t)(k0 + colL) * HDd + 32 + bk16);
      v16h kf10 = *(const v16h*)(kp + (size_t)(k0 + 16 + colL) * HDd + 0  + bk16);
      v16h kf11 = *(const v16h*)(kp + (size_t)(k0 + 16 + colL) * HDd + 32 + bk16);
      s0 = __builtin_amdgcn_wmma_f32_16x16x32_f16(false, qf0, false, kf00, (short)0, s0, false, false);
      s0 = __builtin_amdgcn_wmma_f32_16x16x32_f16(false, qf1, false, kf01, (short)0, s0, false, false);
      s1 = __builtin_amdgcn_wmma_f32_16x16x32_f16(false, qf0, false, kf10, (short)0, s1, false, false);
      s1 = __builtin_amdgcn_wmma_f32_16x16x32_f16(false, qf1, false, kf11, (short)0, s1, false, false);
    }

    const int key0 = k0 + colL;
    const int key1 = key0 + 16;
    const float ma0 = mrow[key0];
    const float ma1 = mrow[key1];

#pragma unroll
    for (int r = 0; r < 8; ++r) {
      const int qi = q0 + r + rhalf;
      float e0 = s0[r] * 0.125f + ma0; if (key0 > qi) e0 = -30000.f;
      float e1 = s1[r] * 0.125f + ma1; if (key1 > qi) e1 = -30000.f;
      // 16-lane row-max (rows live in lane halves; xor<16 stays in-half)
      float v = fmaxf(e0, e1);
      v = fmaxf(v, __shfl_xor(v, 1, 32));
      v = fmaxf(v, __shfl_xor(v, 2, 32));
      v = fmaxf(v, __shfl_xor(v, 4, 32));
      v = fmaxf(v, __shfl_xor(v, 8, 32));
      const float mnew = fmaxf(rmax[r], v);
      const float corr = __expf(rmax[r] - mnew);
      rmax[r] = mnew;
      const float p0 = __expf(e0 - mnew);
      const float p1 = __expf(e1 - mnew);
      float ps = p0 + p1;
      ps += __shfl_xor(ps, 1, 32);
      ps += __shfl_xor(ps, 2, 32);
      ps += __shfl_xor(ps, 4, 32);
      ps += __shfl_xor(ps, 8, 32);
      rsum[r] = rsum[r] * corr + ps;
#pragma unroll
      for (int dt = 0; dt < 4; ++dt) accO[dt][r] *= corr;
      // stash P (C-layout) into per-wave LDS tile
      const int m = r + rhalf;
      myld[m * 40 + colL]      = (f16)p0;
      myld[m * 40 + 16 + colL] = (f16)p1;
    }
    asm volatile("" ::: "memory");   // compiler barrier; same-wave DS ops are in-order

    // re-read P as an A-fragment (16x32)
    const f16* lp = myld + (size_t)colL * 40 + ((lane >> 4) * 16);
    v8h plo = *(const v8h*)lp;
    v8h phi = *(const v8h*)(lp + 8);
    v16h pf = make_frag(plo, phi);

#pragma unroll
    for (int dt = 0; dt < 4; ++dt) {
      // V B-fragment: col N = d, 16 contiguous keys from transposed v
      v16h vf = *(const v16h*)(vp + (size_t)(dt * 16 + colL) * Sq + k0 + bk16);
      accO[dt] = __builtin_amdgcn_wmma_f32_16x16x32_f16(false, pf, false, vf,
                                                        (short)0, accO[dt], false, false);
    }
    asm volatile("" ::: "memory");
  }

#pragma unroll
  for (int dt = 0; dt < 4; ++dt) {
#pragma unroll
    for (int r = 0; r < 8; ++r) {
      const int m = q0 + r + rhalf;
      const int d = dt * 16 + colL;
      out[((size_t)b * Sq + m) * Hd + h * HDd + d] = accO[dt][r] / rsum[r];
    }
  }
}

// ---------------- launcher ----------------
extern "C" void kernel_launch(void* const* d_in, const int* in_sizes, int n_in,
                              void* d_out, int out_size, void* d_ws, size_t ws_size,
                              hipStream_t stream) {
  const float* x    = (const float*)d_in[0];
  const float* mask = (const float*)d_in[1];
  const float* Wq = (const float*)d_in[2];
  const float* bq = (const float*)d_in[3];
  const float* Aq = (const float*)d_in[4];
  const float* Bq = (const float*)d_in[5];
  const float* Wk = (const float*)d_in[6];
  const float* bk = (const float*)d_in[7];
  const float* Ak = (const float*)d_in[8];
  const float* Bk = (const float*)d_in[9];
  const float* Wv = (const float*)d_in[10];
  const float* bv = (const float*)d_in[11];
  const float* Av = (const float*)d_in[12];
  const float* Bv = (const float*)d_in[13];

  char* ws = (char*)d_ws;
  f16*   Xh = (f16*)ws;   ws += (size_t)Md * Hd * sizeof(f16);
  f16*   Wh = (f16*)ws;   ws += (size_t)3 * Hd * Hd * sizeof(f16);
  float* T  = (float*)ws; ws += (size_t)3 * Md * 4 * sizeof(float);
  f16*   qh = (f16*)ws;   ws += (size_t)Md * Hd * sizeof(f16);
  f16*   kh = (f16*)ws;   ws += (size_t)Md * Hd * sizeof(f16);
  f16*   vT = (f16*)ws;   ws += (size_t)Md * Hd * sizeof(f16);

  const int nX = Md * Hd;
  cvt_x_kernel<<<nX / 256, 256, 0, stream>>>(x, Xh, nX);
  cvt_w_kernel<<<(3 * Hd * Hd) / 256, 256, 0, stream>>>(Wq, Wk, Wv, Wh);
  lora_t_kernel<<<(3 * Md * 4) / 256, 256, 0, stream>>>(x, Aq, Ak, Av, T);

  qkv_gemm_kernel<<<dim3(Hd / 64, Md / 128, 3), 256, 0, stream>>>(
      Xh, Wh, T, bq, bk, bv, Bq, Bk, Bv, qh, kh, vT);

  attn_kernel<<<dim3(Sq / 64, NHd, Bsz), 128, 0, stream>>>(
      qh, kh, vT, mask, (float*)d_out);
}
